// WDRNNEqualizer_71047349010656
// MI455X (gfx1250) — matrix operand order repaired and checked by reading, not compiled, not run on hardware
//
#include <hip/hip_runtime.h>

// Decision-feedback equalizer (WDRNN), CDNA5 / gfx1250, wave32.
//
// One wave owns 16 sequences. Per step, in transposed orientation
// (hidden units = WMMA rows, sequences = WMMA columns):
//   cw (next step)  = W1a^T x win^T          -- 2x v_wmma, off critical path
//   c  (this step)  = W1b^T x buf^T + cw     -- 2x v_wmma, critical
//   y[m] = sum_n tanh(c[n][m]) * W2[n] + b2  -- 16 fma (4 parallel chains) + 1 shfl_xor(16)
//   PAM-4 decision -> compressed-sigmoid blend -> feedback funnel-shift
//
// All hidden units of sequence m live on lanes {m, m+16}; y lands on the lane
// that owns the feedback buffer, so no broadcasts are needed. b1 rides as a
// fake feedback tap (B_buf[K=15]=1, W1b row15=b1). The x window slides in
// registers (v_alignbit funnel shifts + one cross-half shuffle); fresh x values
// come from a 2-deep prefetch queue so loads never gate the recurrence.
// The loop body is a single basic block (branchless store: lanes m and m+16
// write bit-identical y to the same address).

typedef __attribute__((ext_vector_type(16))) _Float16     v16h;
typedef __attribute__((ext_vector_type(8)))  float        v8f;
typedef __attribute__((ext_vector_type(8)))  unsigned int v8u;

#define ITAPS 31
#define FBTAPS 15
#define HID 32
#define TLEN 32768

__device__ __forceinline__ float fexp2(float x) { return __builtin_amdgcn_exp2f(x); }
__device__ __forceinline__ float frcp(float x)  { return __builtin_amdgcn_rcpf(x); }

__device__ __forceinline__ float tanh_f(float x) {
#if __has_builtin(__builtin_amdgcn_tanhf)
  return __builtin_amdgcn_tanhf(x);            // gfx1250 v_tanh_f32 (trans)
#else
  float e = fexp2(x * 2.8853900817779268f);
  return 1.0f - 2.0f * frcp(1.0f + e);
#endif
}

__device__ __forceinline__ unsigned int funnel16(unsigned int hi32, unsigned int lo32) {
  // (hi32 << 16) | (lo32 >> 16)  ==  v_alignbit_b32 hi32, lo32, 16
#if __has_builtin(__builtin_amdgcn_alignbit)
  return __builtin_amdgcn_alignbit(hi32, lo32, 16);
#else
  return (hi32 << 16) | (lo32 >> 16);
#endif
}

// Feedback buffer: shift halves right by one, insert `ins` at j=0, keep j=15 (bias=1.0h).
__device__ __forceinline__ v16h buf_shift_in(v16h b, unsigned short ins) {
  v8u u = __builtin_bit_cast(v8u, b);
  v8u r;
  r[0] = (u[0] << 16) | (unsigned int)ins;
  #pragma unroll
  for (int v = 1; v < 7; ++v) r[v] = funnel16(u[v], u[v - 1]);
  r[7] = 0x3C000000u | (u[6] >> 16);           // high half = 1.0h bias column
  return __builtin_bit_cast(v16h, r);
}

// Window: shift halves left by one, insert `tail` at j=15.
__device__ __forceinline__ v16h win_shift(v16h w, unsigned short tail) {
  v8u u = __builtin_bit_cast(v8u, w);
  v8u r;
  #pragma unroll
  for (int v = 0; v < 7; ++v) r[v] = funnel16(u[v + 1], u[v]);
  r[7] = (u[7] >> 16) | ((unsigned int)tail << 16);
  return __builtin_bit_cast(v16h, r);
}

__device__ __forceinline__ unsigned short h_bits(_Float16 h) {
  return __builtin_bit_cast(unsigned short, h);
}

__global__ void __launch_bounds__(32) dfe_wmma_kernel(
    const float* __restrict__ x,   // (B, T)
    const float* __restrict__ W1,  // (46, 32) row-major
    const float* __restrict__ b1,  // (32)
    const float* __restrict__ W2,  // (32)
    const float* __restrict__ b2,  // (1)
    float* __restrict__ out,       // (B, T)
    int T)
{
  const int lane = threadIdx.x;        // 0..31
  const int m    = lane & 15;          // sequence column this lane owns
  const int hi   = lane >> 4;          // half-wave (K-range selector)
  const int row  = blockIdx.x * 16 + m;
  const long xrow = (long)row * (long)T;

  // ---- Static A matrices: W1^T in WMMA 16-bit A layout (A[p=n][k=feature]).
  // This code's k order (contiguous, same as B): lane L holds row p = L&15, halves j -> K = j + 16*hi.
  v16h aw_lo, aw_hi, ab_lo, ab_hi;
  #pragma unroll
  for (int j = 0; j < 16; ++j) {
    int k = j + 16 * hi;
    float w0lo = (k < ITAPS) ? W1[k * HID + m]        : 0.0f;
    float w0hi = (k < ITAPS) ? W1[k * HID + (m + 16)] : 0.0f;
    float w1lo = (k < FBTAPS) ? W1[(ITAPS + k) * HID + m]
                              : ((k == FBTAPS) ? b1[m] : 0.0f);
    float w1hi = (k < FBTAPS) ? W1[(ITAPS + k) * HID + (m + 16)]
                              : ((k == FBTAPS) ? b1[m + 16] : 0.0f);
    aw_lo[j] = (_Float16)w0lo;  aw_hi[j] = (_Float16)w0hi;
    ab_lo[j] = (_Float16)w1lo;  ab_hi[j] = (_Float16)w1hi;
  }

  // ---- Second layer weights, per-lane slices matching D's C-layout rows.
  float w2a[8], w2b[8];
  #pragma unroll
  for (int r = 0; r < 8; ++r) {
    w2a[r] = W2[8 * hi + r];          // rows n = r + 8*hi      (c0)
    w2b[r] = W2[16 + 8 * hi + r];     // rows n = 16 + r + 8*hi (c1)
  }
  const float b2v = b2[0];

  // ---- B matrices (lane holds column m; contiguous k order as in A: halves j -> K = j + 16*hi).
  v16h bwin, bbuf;
  #pragma unroll
  for (int j = 0; j < 16; ++j) {
    int xi = (j + 16 * hi) - (ITAPS - 1);               // step-0 x index
    bwin[j] = (xi >= 0) ? (_Float16)x[xrow + xi] : (_Float16)0.0f;
    bbuf[j] = (_Float16)0.0f;
  }
  bbuf[15] = (_Float16)1.0f;   // bias column (pairs with W1b row15 = b1)

  // ---- Prologue: window WMMAs for step 0, 2-deep x prefetch queue.
  const v8f z = {};
  v8f cw0 = __builtin_amdgcn_wmma_f32_16x16x32_f16(false, aw_lo, false, bwin, (short)0, z, false, false);
  v8f cw1 = __builtin_amdgcn_wmma_f32_16x16x32_f16(false, aw_hi, false, bwin, (short)0, z, false, false);
  float q0 = x[xrow + ((2 < T) ? 2 : T - 1)];   // inserted at step 0 (xp[32])
  float q1 = x[xrow + ((3 < T) ? 3 : T - 1)];   // inserted at step 1

  for (int i = 0; i < T; ++i) {
    // ---- critical: feedback matmul accumulated onto pre-computed window part
    v8f c0 = __builtin_amdgcn_wmma_f32_16x16x32_f16(false, ab_lo, false, bbuf, (short)0, cw0, false, false);
    v8f c1 = __builtin_amdgcn_wmma_f32_16x16x32_f16(false, ab_hi, false, bbuf, (short)0, cw1, false, false);

    // ---- off-critical (fills WMMA->VALU hazard slots): slide window, issue
    //      next step's window WMMAs, rotate the x prefetch queue.
    {
      unsigned short head = h_bits(bwin[0]);
      int pr = __shfl_xor((int)head, 16, 32);                 // partner's old j=0 (K=16)
      _Float16 tail_lo = __builtin_bit_cast(_Float16, (unsigned short)pr);
      _Float16 tail_hi = (_Float16)q0;                        // enters dead column K=31
      bwin = win_shift(bwin, h_bits(hi ? tail_hi : tail_lo));
    }
    cw0 = __builtin_amdgcn_wmma_f32_16x16x32_f16(false, aw_lo, false, bwin, (short)0, z, false, false);
    cw1 = __builtin_amdgcn_wmma_f32_16x16x32_f16(false, aw_hi, false, bwin, (short)0, z, false, false);
    q0 = q1;
    {
      int nxt = i + 4;                                        // x index for step i+2 insert
      if (nxt > T - 1) nxt = T - 1;                           // clamped loads feed dead columns only
      q1 = x[xrow + nxt];
    }

    // ---- y[m] = sum_n tanh(h[n][m]) * W2[n] + b2 ; 4 parallel fma chains
    float p0 = 0.0f, p1 = 0.0f, p2 = 0.0f, p3 = 0.0f;
    #pragma unroll
    for (int r = 0; r < 8; r += 2) {
      p0 = fmaf(tanh_f(c0[r]),     w2a[r],     p0);
      p1 = fmaf(tanh_f(c0[r + 1]), w2a[r + 1], p1);
      p2 = fmaf(tanh_f(c1[r]),     w2b[r],     p2);
      p3 = fmaf(tanh_f(c1[r + 1]), w2b[r + 1], p3);
    }
    float p = (p0 + p1) + (p2 + p3);
    float y = p + __shfl_xor(p, 16, 32) + b2v;   // combine the two half-wave partials

    // Branchless store: lanes m and m+16 hold bit-identical y (commutative sum)
    // and the same address, so a duplicate-lane write is deterministic. Keeps
    // the loop body a single basic block for the scheduler.
    out[xrow + i] = y;

    // ---- PAM-4 decision + compressed sigmoid blend (critical)
    float yc    = fminf(fmaxf(y, -3.0f), 3.0f);
    float yhat  = 2.0f * floorf(yc * 0.5f) + 1.0f;
    float gamma = 1.0f - fabsf(y - yhat);
    float v     = 8.0f * gamma - 4.0f;           // ALPHA*(gamma/BETA - 1)
    float s     = frcp(1.0f + fexp2(-1.4426950408889634f * v));
    float yt    = fmaf(s, yhat - y, y);

    // ---- feedback shift: 8 funnel shifts instead of 15 b16 moves (critical)
    bbuf = buf_shift_in(bbuf, h_bits((_Float16)yt));
  }
}

extern "C" void kernel_launch(void* const* d_in, const int* in_sizes, int n_in,
                              void* d_out, int out_size, void* d_ws, size_t ws_size,
                              hipStream_t stream) {
  const float* x  = (const float*)d_in[0];
  const float* W1 = (const float*)d_in[1];
  const float* b1 = (const float*)d_in[2];
  const float* W2 = (const float*)d_in[3];
  const float* b2 = (const float*)d_in[4];
  float* out = (float*)d_out;

  const int T = TLEN;
  const int B = in_sizes[0] / T;     // 512
  dim3 grid(B / 16), block(32);      // one wave32 per 16 sequences
  dfe_wmma_kernel<<<grid, block, 0, stream>>>(x, W1, b1, W2, b2, out, T);
}
